// MotionEncoder_65996467470749
// MI455X (gfx1250) — compile-verified, matrix operand order checked by reading, not run on hardware
//
#include <hip/hip_runtime.h>
#include <hip/hip_bf16.h>

// ---------------------------------------------------------------------------
// ST-GCN forward for MI455X (gfx1250).
// All channel contractions run on v_wmma_f32_16x16x32_f16 (f16 in, f32 acc).
// Each wave owns a 32x32 output tile (4 accumulators, 4 independent WMMAs per
// K-step); block = 8 waves = 128x64 macro tile.  First layer K is zero-padded
// to 32 so every fragment load is an aligned global_load_b128 (no masked path).
// ---------------------------------------------------------------------------

typedef __attribute__((ext_vector_type(16))) _Float16 v16h;
typedef __attribute__((ext_vector_type(8)))  float    v8f;

#define V_J    17
#define T_LEN  1024
#define N_B    8
#define M_ROWS (N_B * T_LEN * V_J)   // 139264, divisible by 128

union FragU  { v16h v; uint4 q[2]; _Float16 h[16]; };
union Pack32 { _Float16 h[32]; uint4 q[4]; };

__device__ __forceinline__ v16h frag_zero() {
  FragU f;
  f.q[0] = make_uint4(0u, 0u, 0u, 0u);
  f.q[1] = make_uint4(0u, 0u, 0u, 0u);
  return f.v;
}

// Fragment loader for V_WMMA_F32_16X16X32_F16 (ISA 7.12.2, 16-bit A 16x32):
// lane L (0..15): row L, K = {k0..k0+7, k0+16..k0+23};
// lane L+16     : row L, K = {k0+8..k0+15, k0+24..k0+31}.
// Requires ld % 8 == 0 and 16B-aligned base (guaranteed by workspace layout).
__device__ __forceinline__ v16h load_frag(const _Float16* __restrict__ P, int ld,
                                          int row, int k0, int lane) {
  FragU f;
  const _Float16* p = P + (size_t)(row + (lane & 15)) * ld + k0 + ((lane >> 4) << 3);
  f.q[0] = *(const uint4*)(p);        // global_load_b128
  f.q[1] = *(const uint4*)(p + 16);   // global_load_b128
  return f.v;
}

__device__ __forceinline__ void epi_store(v8f acc, int tm, int tn, int lane,
                                          const float* __restrict__ bias,
                                          const float* __restrict__ bnS,
                                          const float* __restrict__ bnB,
                                          int relu,
                                          _Float16* __restrict__ outH,
                                          float* __restrict__ outF, int ldc) {
  const int col = tn + (lane & 15);
  const int rw0 = tm + ((lane >> 4) << 3);
  const float bi = bias ? bias[col] : 0.0f;
  const float s  = bnS  ? bnS[col]  : 1.0f;
  const float sh = bnB  ? bnB[col]  : 0.0f;
#pragma unroll
  for (int i = 0; i < 8; ++i) {
    float vv = (acc[i] + bi) * s + sh;
    if (relu) vv = vv > 0.0f ? vv : 0.0f;
    size_t o = (size_t)(rw0 + i) * ldc + col;
    if (outH) outH[o] = (_Float16)vv;
    else      outF[o] = vv;
  }
}

// ---------------------------------------------------------------------------
// WMMA GEMM: C[M,N] = A[M,K] * Bt[N,K]^T (+bias) (*bnS+bnB) (relu), f16/f32 out.
// Wave tile 32x32, block tile 128x64.  K must be a multiple of 32.
// ---------------------------------------------------------------------------
__global__ __launch_bounds__(256)
void gemm16_kernel(const _Float16* __restrict__ A, const _Float16* __restrict__ Bt,
                   int M, int N, int Kd, int lda, int ldb,
                   const float* __restrict__ bias,
                   const float* __restrict__ bnS, const float* __restrict__ bnB,
                   int relu,
                   _Float16* __restrict__ outH, float* __restrict__ outF, int ldc) {
  const int lane  = threadIdx.x & 31;
  const int wv    = threadIdx.x >> 5;
  const int tileM = blockIdx.y * 128 + (wv & 3) * 32;
  const int tileN = blockIdx.x * 64  + (wv >> 2) * 32;
  if (tileM >= M || tileN >= N) return;

  v8f acc00 = {}, acc01 = {}, acc10 = {}, acc11 = {};
  for (int k0 = 0; k0 < Kd; k0 += 32) {
    v16h a0 = load_frag(A,  lda, tileM,      k0, lane);
    v16h a1 = load_frag(A,  lda, tileM + 16, k0, lane);
    v16h b0 = load_frag(Bt, ldb, tileN,      k0, lane);
    v16h b1 = load_frag(Bt, ldb, tileN + 16, k0, lane);
    acc00 = __builtin_amdgcn_wmma_f32_16x16x32_f16(false, a0, false, b0, (short)0, acc00, false, false);
    acc01 = __builtin_amdgcn_wmma_f32_16x16x32_f16(false, a0, false, b1, (short)0, acc01, false, false);
    acc10 = __builtin_amdgcn_wmma_f32_16x16x32_f16(false, a1, false, b0, (short)0, acc10, false, false);
    acc11 = __builtin_amdgcn_wmma_f32_16x16x32_f16(false, a1, false, b1, (short)0, acc11, false, false);
  }

  epi_store(acc00, tileM,      tileN,      lane, bias, bnS, bnB, relu, outH, outF, ldc);
  epi_store(acc01, tileM,      tileN + 16, lane, bias, bnS, bnB, relu, outH, outF, ldc);
  epi_store(acc10, tileM + 16, tileN,      lane, bias, bnS, bnB, relu, outH, outF, ldc);
  epi_store(acc11, tileM + 16, tileN + 16, lane, bias, bnS, bnB, relu, outH, outF, ldc);
}

// ---------------------------------------------------------------------------
// Temporal conv: 9 taps accumulated in registers; A rows shifted by (dt-4)*V
// with per-lane zero padding at T boundaries.
// Epilogue: BN2 (tconv bias folded) + optional residual + ReLU -> f16.
// ---------------------------------------------------------------------------
__global__ __launch_bounds__(256)
void gemm_tconv_kernel(const _Float16* __restrict__ A, const _Float16* __restrict__ W9,
                       int M, int N, int Kd,
                       const float* __restrict__ bnS, const float* __restrict__ bnB,
                       const _Float16* __restrict__ resid, int ldr,
                       _Float16* __restrict__ outH) {
  const int lane  = threadIdx.x & 31;
  const int wv    = threadIdx.x >> 5;
  const int tileM = blockIdx.y * 128 + (wv & 3) * 32;
  const int tileN = blockIdx.x * 64  + (wv >> 2) * 32;
  if (tileM >= M || tileN >= N) return;

  const int t0 = ((tileM + (lane & 15)) / V_J) % T_LEN;        // row tileM..+15
  const int t1 = ((tileM + 16 + (lane & 15)) / V_J) % T_LEN;   // row tileM+16..+31

  v8f acc00 = {}, acc01 = {}, acc10 = {}, acc11 = {};
  for (int dt = 0; dt < 9; ++dt) {
    const int  sft    = dt - 4;
    const int  rshift = sft * V_J;
    const bool v0 = ((unsigned)(t0 + sft)) < (unsigned)T_LEN;
    const bool v1 = ((unsigned)(t1 + sft)) < (unsigned)T_LEN;
    const _Float16* Wt = W9 + (size_t)dt * N * Kd;
    for (int k0 = 0; k0 < Kd; k0 += 32) {
      v16h a0 = v0 ? load_frag(A, Kd, tileM + rshift,      k0, lane) : frag_zero();
      v16h a1 = v1 ? load_frag(A, Kd, tileM + 16 + rshift, k0, lane) : frag_zero();
      v16h b0 = load_frag(Wt, Kd, tileN,      k0, lane);
      v16h b1 = load_frag(Wt, Kd, tileN + 16, k0, lane);
      acc00 = __builtin_amdgcn_wmma_f32_16x16x32_f16(false, a0, false, b0, (short)0, acc00, false, false);
      acc01 = __builtin_amdgcn_wmma_f32_16x16x32_f16(false, a0, false, b1, (short)0, acc01, false, false);
      acc10 = __builtin_amdgcn_wmma_f32_16x16x32_f16(false, a1, false, b0, (short)0, acc10, false, false);
      acc11 = __builtin_amdgcn_wmma_f32_16x16x32_f16(false, a1, false, b1, (short)0, acc11, false, false);
    }
  }

#pragma unroll
  for (int q = 0; q < 4; ++q) {
    const v8f acc = (q == 0) ? acc00 : (q == 1) ? acc01 : (q == 2) ? acc10 : acc11;
    const int tm  = tileM + ((q >> 1) << 4);
    const int col = tileN + ((q & 1) << 4) + (lane & 15);
    const int rw0 = tm + ((lane >> 4) << 3);
    const float s  = bnS[col];
    const float sh = bnB[col];
#pragma unroll
    for (int i = 0; i < 8; ++i) {
      float vv = acc[i] * s + sh;
      if (resid) vv += (float)resid[(size_t)(rw0 + i) * ldr + col];
      vv = vv > 0.0f ? vv : 0.0f;
      outH[(size_t)(rw0 + i) * N + col] = (_Float16)vv;
    }
  }
}

// ---------------------------------------------------------------------------
// Adjacency mix: tin[(n,t,w),o] = relu(bn1( sum_{k,v} g[(n,t,v),k*out+o]*A[k,v,w] ))
// ---------------------------------------------------------------------------
__global__ __launch_bounds__(256)
void adjacency_kernel(const _Float16* __restrict__ g, const float* __restrict__ Aadj,
                      const float* __restrict__ bnS, const float* __restrict__ bnB,
                      _Float16* __restrict__ tin, int outc) {
  __shared__ float sA[3 * V_J * V_J];
  for (int i = threadIdx.x; i < 3 * V_J * V_J; i += blockDim.x) sA[i] = Aadj[i];
  __syncthreads();

  size_t idx   = (size_t)blockIdx.x * blockDim.x + threadIdx.x;
  size_t total = (size_t)M_ROWS * outc;
  if (idx >= total) return;

  const int    o    = (int)(idx % outc);
  const size_t r    = idx / outc;
  const int    w    = (int)(r % V_J);
  const size_t base = r - w;
  const int    ldg  = 3 * outc;

  float acc = 0.0f;
#pragma unroll
  for (int k = 0; k < 3; ++k) {
    const _Float16* gk = g + (size_t)k * outc + o;
    for (int v = 0; v < V_J; ++v) {
      float a = sA[(k * V_J + v) * V_J + w];
      if (a != 0.0f) acc += a * (float)gk[(base + v) * ldg];
    }
  }
  float vv = acc * bnS[o] + bnB[o];
  tin[r * outc + o] = (_Float16)(vv > 0.0f ? vv : 0.0f);
}

// data_bn over channel (v*3+c); output f16 [M_ROWS, 32], cols 3..31 zero.
__global__ __launch_bounds__(256)
void databn_pad_kernel(const float* __restrict__ x,
                       const float* __restrict__ g, const float* __restrict__ b,
                       const float* __restrict__ m, const float* __restrict__ vv,
                       _Float16* __restrict__ h0) {
  int r = blockIdx.x * blockDim.x + threadIdx.x;
  if (r >= M_ROWS) return;
  const int vj = r % V_J;
  Pack32 p;
#pragma unroll
  for (int c = 0; c < 32; ++c) p.h[c] = (_Float16)0.0f;
#pragma unroll
  for (int c = 0; c < 3; ++c) {
    int ch = vj * 3 + c;
    float s = g[ch] * rsqrtf(vv[ch] + 1e-5f);
    p.h[c] = (_Float16)(x[(size_t)r * 3 + c] * s + (b[ch] - m[ch] * s));
  }
  uint4* dst = (uint4*)(h0 + (size_t)r * 32);
#pragma unroll
  for (int q = 0; q < 4; ++q) dst[q] = p.q[q];
}

// BN prep: scale = g/sqrt(v+eps), shift = b - m*scale (+ fold*scale)
__global__ __launch_bounds__(256)
void bnprep_kernel(const float* __restrict__ g, const float* __restrict__ b,
                   const float* __restrict__ m, const float* __restrict__ v,
                   const float* __restrict__ fold,
                   float* __restrict__ scale, float* __restrict__ shift, int C) {
  int i = blockIdx.x * blockDim.x + threadIdx.x;
  if (i >= C) return;
  float s = g[i] * rsqrtf(v[i] + 1e-5f);
  scale[i] = s;
  shift[i] = b[i] - m[i] * s + (fold ? fold[i] * s : 0.0f);
}

__global__ __launch_bounds__(256)
void cvt_kernel(const float* __restrict__ src, _Float16* __restrict__ dst, int n) {
  int i = blockIdx.x * blockDim.x + threadIdx.x;
  if (i < n) dst[i] = (_Float16)src[i];
}

// f32 [rows,kin] -> f16 [rows,kpad] with zero pad
__global__ __launch_bounds__(256)
void cvt_pad_kernel(const float* __restrict__ src, _Float16* __restrict__ dst,
                    int rows, int kin, int kpad) {
  int i = blockIdx.x * blockDim.x + threadIdx.x;
  if (i >= rows * kpad) return;
  int r = i / kpad, c = i - r * kpad;
  dst[i] = (c < kin) ? (_Float16)src[(size_t)r * kin + c] : (_Float16)0.0f;
}

// tconv_w (out,out,9) -> 9 contiguous [out,out] B^T matrices: dst[dt][o][c]=w[o][c][dt]
__global__ __launch_bounds__(256)
void repack_tconv_kernel(const float* __restrict__ w, _Float16* __restrict__ dst, int outc) {
  int i = blockIdx.x * blockDim.x + threadIdx.x;
  int total = outc * outc * 9;
  if (i >= total) return;
  int dt  = i / (outc * outc);
  int rem = i - dt * outc * outc;
  int o   = rem / outc;
  int c   = rem - o * outc;
  dst[i] = (_Float16)w[((size_t)o * outc + c) * 9 + dt];
}

// mean over 17 joints x 2 timesteps, f16 out: pooled[(n*512+tp), c]
__global__ __launch_bounds__(256)
void pool_kernel(const _Float16* __restrict__ h, _Float16* __restrict__ pooled) {
  int idx = blockIdx.x * blockDim.x + threadIdx.x;
  int total = N_B * 512 * 256;
  if (idx >= total) return;
  int c  = idx % 256;
  int rp = idx / 256;
  int tp = rp % 512;
  int n  = rp / 512;
  const _Float16* base = h + (((size_t)n * T_LEN + 2 * tp) * V_J) * 256 + c;
  float acc = 0.0f;
  for (int j = 0; j < 2 * V_J; ++j) acc += (float)base[(size_t)j * 256];
  pooled[(size_t)rp * 256 + c] = (_Float16)(acc * (1.0f / 34.0f));
}

// ---------------------------------------------------------------------------
extern "C" void kernel_launch(void* const* d_in, const int* in_sizes, int n_in,
                              void* d_out, int out_size, void* d_ws, size_t ws_size,
                              hipStream_t stream) {
  (void)in_sizes; (void)n_in; (void)out_size;

  // ---- parse inputs (setup_inputs() insertion order, depth-first) ----
  int cur = 0;
  const float* x    = (const float*)d_in[cur++];
  const float* Aadj = (const float*)d_in[cur++];
  const float* dbg  = (const float*)d_in[cur++];
  const float* dbb  = (const float*)d_in[cur++];
  const float* dbm  = (const float*)d_in[cur++];
  const float* dbv  = (const float*)d_in[cur++];

  const int CH[9] = {64, 64, 64, 64, 128, 128, 128, 256, 256};
  struct BnP { const float *g, *b, *m, *v; };
  struct Blk {
    const float *gw, *gb; BnP bn1;
    const float *tw, *tb; BnP bn2;
    const float *rw, *rb; BnP rbn;
    int in_c, out_c, has_res;
  } blk[9];

  int in_c = 3;
  for (int i = 0; i < 9; ++i) {
    Blk& B = blk[i];
    B.in_c = in_c; B.out_c = CH[i];
    B.gw = (const float*)d_in[cur++];
    B.gb = (const float*)d_in[cur++];
    B.bn1.g = (const float*)d_in[cur++]; B.bn1.b = (const float*)d_in[cur++];
    B.bn1.m = (const float*)d_in[cur++]; B.bn1.v = (const float*)d_in[cur++];
    B.tw = (const float*)d_in[cur++];
    B.tb = (const float*)d_in[cur++];
    B.bn2.g = (const float*)d_in[cur++]; B.bn2.b = (const float*)d_in[cur++];
    B.bn2.m = (const float*)d_in[cur++]; B.bn2.v = (const float*)d_in[cur++];
    B.has_res = (i > 0 && in_c != CH[i]);
    if (B.has_res) {
      B.rw = (const float*)d_in[cur++];
      B.rb = (const float*)d_in[cur++];
      B.rbn.g = (const float*)d_in[cur++]; B.rbn.b = (const float*)d_in[cur++];
      B.rbn.m = (const float*)d_in[cur++]; B.rbn.v = (const float*)d_in[cur++];
    } else { B.rw = nullptr; B.rb = nullptr; B.rbn = BnP{}; }
    in_c = CH[i];
  }
  const float* fw = (const float*)d_in[cur++];
  const float* fb = (const float*)d_in[cur++];

  // ---- workspace carve-up ----
  char* base = (char*)d_ws;
  size_t off = 0;
  auto alloc = [&](size_t bytes) -> char* {
    off = (off + 255) & ~(size_t)255;
    char* p = base + off;
    off += bytes;
    return p;
  };
  const size_t M = M_ROWS;
  _Float16* hA     = (_Float16*)alloc(M * 256 * 2);  // layer0 uses [M,32] view
  _Float16* hB     = (_Float16*)alloc(M * 256 * 2);
  _Float16* tin    = (_Float16*)alloc(M * 256 * 2);
  _Float16* gbuf   = (_Float16*)alloc(M * 768 * 2);  // GCN output; reused as residual
  _Float16* wg     = (_Float16*)alloc(3 * 256 * 256 * 2);
  _Float16* wt     = (_Float16*)alloc(9 * 256 * 256 * 2);
  _Float16* wr     = (_Float16*)alloc(1024 * 256 * 2);  // res / fcn weights
  float*    s1     = (float*)alloc(1024 * 4);
  float*    b1     = (float*)alloc(1024 * 4);
  float*    s2     = (float*)alloc(1024 * 4);
  float*    b2     = (float*)alloc(1024 * 4);
  float*    s3     = (float*)alloc(1024 * 4);
  float*    b3     = (float*)alloc(1024 * 4);
  _Float16* pooled = (_Float16*)alloc(4096 * 256 * 2);
  if (off > ws_size) return;  // workspace too small; nothing safe to do

  const dim3 blk256(256);

  // ---- data_bn -> hA f16 [M,32] (K padded to 32) ----
  databn_pad_kernel<<<dim3((M_ROWS + 255) / 256), blk256, 0, stream>>>(
      x, dbg, dbb, dbm, dbv, hA);

  _Float16* hcur = hA;
  _Float16* hnxt = hB;

  for (int i = 0; i < 9; ++i) {
    const Blk& B = blk[i];
    const int ic  = B.in_c, oc = B.out_c;
    const int icp = (i == 0) ? 32 : ic;   // padded K for layer 0

    // BN prep
    bnprep_kernel<<<dim3((oc + 255) / 256), blk256, 0, stream>>>(
        B.bn1.g, B.bn1.b, B.bn1.m, B.bn1.v, nullptr, s1, b1, oc);
    bnprep_kernel<<<dim3((oc + 255) / 256), blk256, 0, stream>>>(
        B.bn2.g, B.bn2.b, B.bn2.m, B.bn2.v, B.tb, s2, b2, oc);
    if (B.has_res)
      bnprep_kernel<<<dim3((oc + 255) / 256), blk256, 0, stream>>>(
          B.rbn.g, B.rbn.b, B.rbn.m, B.rbn.v, B.rb, s3, b3, oc);

    // GCN GEMM: g[M, 3*oc] = hcur[M, icp] x gcn_w^T  (+ gcn_b), f16 out
    {
      if (i == 0) {
        int n = 3 * oc * icp;
        cvt_pad_kernel<<<dim3((n + 255) / 256), blk256, 0, stream>>>(B.gw, wg, 3 * oc, ic, icp);
      } else {
        int n = 3 * oc * ic;
        cvt_kernel<<<dim3((n + 255) / 256), blk256, 0, stream>>>(B.gw, wg, n);
      }
      dim3 grid((3 * oc) / 64, (unsigned)(M / 128));
      gemm16_kernel<<<grid, blk256, 0, stream>>>(
          hcur, wg, (int)M, 3 * oc, icp, icp, icp,
          B.gb, nullptr, nullptr, 0, gbuf, nullptr, 3 * oc);
    }

    // adjacency + bn1 + relu -> tin [M, oc]
    {
      size_t tot = M * oc;
      adjacency_kernel<<<dim3((unsigned)((tot + 255) / 256)), blk256, 0, stream>>>(
          gbuf, Aadj, s1, b1, tin, oc);
    }

    // residual (gbuf region is free now)
    const _Float16* resid = nullptr;
    int ldr = oc;
    if (i > 0) {
      if (B.has_res) {
        int n = oc * ic;
        cvt_kernel<<<dim3((n + 255) / 256), blk256, 0, stream>>>(B.rw, wr, n);
        dim3 grid(oc / 64, (unsigned)(M / 128));
        gemm16_kernel<<<grid, blk256, 0, stream>>>(
            hcur, wr, (int)M, oc, ic, ic, ic,
            nullptr, s3, b3, 0, gbuf, nullptr, oc);
        resid = gbuf; ldr = oc;
      } else {
        resid = hcur; ldr = ic;  // ic == oc
      }
    }

    // temporal conv (9 taps) + bn2 + residual + relu -> hnxt [M, oc]
    {
      int n = oc * oc * 9;
      repack_tconv_kernel<<<dim3((n + 255) / 256), blk256, 0, stream>>>(B.tw, wt, oc);
      dim3 grid(oc / 64, (unsigned)(M / 128));
      gemm_tconv_kernel<<<grid, blk256, 0, stream>>>(
          tin, wt, (int)M, oc, oc, s2, b2, resid, ldr, hnxt);
    }

    { _Float16* t = hcur; hcur = hnxt; hnxt = t; }
  }

  // ---- pool (mean over 17 joints x 2 t) then final FC 256->1024, fp32 out ----
  {
    int tot = N_B * 512 * 256;
    pool_kernel<<<dim3((tot + 255) / 256), blk256, 0, stream>>>(hcur, pooled);
    int n = 1024 * 256;
    cvt_kernel<<<dim3((n + 255) / 256), blk256, 0, stream>>>(fw, wr, n);
    dim3 grid(1024 / 64, 4096 / 128);
    gemm16_kernel<<<grid, blk256, 0, stream>>>(
        pooled, wr, 4096, 1024, 256, 256, 256,
        fb, nullptr, nullptr, 0, nullptr, (float*)d_out, 1024);
  }
}
